// GRU_35759897707254
// MI455X (gfx1250) — compile-verified, matrix operand order checked by reading, not run on hardware
//
#include <hip/hip_runtime.h>
#include <hip/hip_bf16.h>
#include <math.h>

typedef __attribute__((ext_vector_type(16))) _Float16 v16h;
typedef __attribute__((ext_vector_type(8)))  _Float16 v8h;
typedef __attribute__((ext_vector_type(8)))  float    v8f;

#define T_STEPS 256
#define BATCH   64
#define DIN     512
#define HENC    1024
#define CH      2048
#define G3      6144   // 3*CH
#define DOUT    512
#define TB      (T_STEPS*BATCH)   // 16384 rows for the batched GEMMs

// Block GEMM tiling
#define BM 64
#define BN 128
#define BK 32

// ---------------------------------------------------------------------------
// CDNA5 async global->LDS copy (ASYNCcnt-tracked) + wait.
//   global_load_async_to_lds_b128  vdst(LDS byte offset), vaddr(64b), off
// ---------------------------------------------------------------------------
__device__ __forceinline__ void async_b128(unsigned lds_off, const _Float16* g) {
  asm volatile("global_load_async_to_lds_b128 %0, %1, off"
               :: "v"(lds_off), "v"(g) : "memory");
}
__device__ __forceinline__ void wait_async0() {
  asm volatile("s_wait_asynccnt 0x0" ::: "memory");
}

// ---------------------------------------------------------------------------
// Fragment mapping for wave32 WMMA f16 (A 16x32 row-major / B = W[N,K] row-major):
// lanes 0-15: row = lane, K chunks {k0..k0+7, k0+16..k0+23};
// lanes 16-31: row = lane-16, K chunks {k0+8..k0+15, k0+24..k0+31}.
// ---------------------------------------------------------------------------
__device__ __forceinline__ v16h load_frag16(const _Float16* __restrict__ base,
                                            int row0, int stride, int k0, int lane) {
  const int l  = lane & 15;
  const int hi = lane >> 4;
  const _Float16* p = base + (size_t)(row0 + l) * (size_t)stride + k0 + hi * 8;
  v8h a = *(const v8h*)(p);
  v8h b = *(const v8h*)(p + 16);
  return __builtin_shufflevector(a, b, 0,1,2,3,4,5,6,7,8,9,10,11,12,13,14,15);
}

// Same mapping, reading from an LDS tile with row stride BK (k0 == 0 inside tile).
__device__ __forceinline__ v16h lds_frag16(const _Float16* tile, int row0, int lane) {
  const int l  = lane & 15;
  const int hi = lane >> 4;
  const _Float16* p = tile + (row0 + l) * BK + hi * 8;
  v8h a = *(const v8h*)(p);
  v8h b = *(const v8h*)(p + 16);
  return __builtin_shufflevector(a, b, 0,1,2,3,4,5,6,7,8,9,10,11,12,13,14,15);
}

__device__ __forceinline__ float act_elu(float x) {
  return x > 0.f ? x : (__expf(x) - 1.f);
}
__device__ __forceinline__ float act_softplus01(float x) {
  return (x > 20.f ? x : log1pf(__expf(x))) + 0.1f;
}
__device__ __forceinline__ float sigmoid_(float x) {
  return 1.f / (1.f + __expf(-x));
}

// ---------------------------------------------------------------------------
__global__ void k_f32_to_f16(const float* __restrict__ in,
                             _Float16* __restrict__ out, int n) {
  int i = blockIdx.x * blockDim.x + threadIdx.x;
  if (i < n) out[i] = (_Float16)in[i];
}

__global__ void k_zero_f16(_Float16* __restrict__ p, int n) {
  int i = blockIdx.x * blockDim.x + threadIdx.x;
  if (i < n) p[i] = (_Float16)0.f;
}

// ---------------------------------------------------------------------------
// WMMA GEMM with double-buffered async-LDS staging:
//   C[M,N] = act( A[M,K] * W[N,K]^T + bias[N] )
// Block: 256 threads = 8 waves (4 M-tiles x 2 N-groups of 64) over a 64x128
// C tile.  Per K-step of 32 the block stages A(64x32)+W(128x32) = 12 KB into
// LDS with global_load_async_to_lds_b128, then 32 WMMAs run from ds_load.
// act: 0 = none, 1 = ELU, 2 = softplus + 0.1
// ---------------------------------------------------------------------------
__global__ void k_gemm_wmma(const _Float16* __restrict__ A,   // [M,K]
                            const _Float16* __restrict__ W,   // [N,K]
                            const float*    __restrict__ bias,// [N]
                            float*          __restrict__ outF32,
                            _Float16*       __restrict__ outF16,
                            int M, int N, int K, int act) {
  __shared__ alignas(16) _Float16 smA[2][BM * BK];   // 2 x 4 KB
  __shared__ alignas(16) _Float16 smB[2][BN * BK];   // 2 x 8 KB

  const int tid  = threadIdx.x;
  const int lane = tid & 31;
  const int wave = tid >> 5;
  const int wm   = wave & 3;        // M sub-tile
  const int wn   = wave >> 2;       // N group (0..1)
  const int m0b  = blockIdx.y * BM;
  const int n0b  = blockIdx.x * BN;

  const unsigned ldsA0 = (unsigned)(uintptr_t)&smA[0][0];
  const unsigned ldsB0 = (unsigned)(uintptr_t)&smB[0][0];

  // Stage one K-tile into buffer `buf` (each thread: 1 A chunk + 2 W chunks).
  auto stage = [&](int buf, int k) {
    {
      const int row = tid >> 2, c = tid & 3;            // 256 chunks of A
      const _Float16* g = A + (size_t)(m0b + row) * K + k + c * 8;
      async_b128(ldsA0 + (unsigned)buf * (BM * BK * 2) + (unsigned)tid * 16u, g);
    }
#pragma unroll
    for (int s = 0; s < 2; ++s) {                       // 512 chunks of W
      const int j = tid + s * 256;
      const int row = j >> 2, c = j & 3;
      const _Float16* g = W + (size_t)(n0b + row) * K + k + c * 8;
      async_b128(ldsB0 + (unsigned)buf * (BN * BK * 2) + (unsigned)j * 16u, g);
    }
  };

  v8f acc0 = {}, acc1 = {}, acc2 = {}, acc3 = {};

  int buf = 0;
  stage(buf, 0);
  wait_async0();
  __syncthreads();

  for (int k = 0; k < K; k += BK) {
    const int nk = k + BK;
    if (nk < K) stage(buf ^ 1, nk);

    v16h a  = lds_frag16(&smA[buf][0], wm * 16, lane);
    v16h b0 = lds_frag16(&smB[buf][0], wn * 64 +  0, lane);
    v16h b1 = lds_frag16(&smB[buf][0], wn * 64 + 16, lane);
    v16h b2 = lds_frag16(&smB[buf][0], wn * 64 + 32, lane);
    v16h b3 = lds_frag16(&smB[buf][0], wn * 64 + 48, lane);
    acc0 = __builtin_amdgcn_wmma_f32_16x16x32_f16(false, a, false, b0, (short)0, acc0, false, false);
    acc1 = __builtin_amdgcn_wmma_f32_16x16x32_f16(false, a, false, b1, (short)0, acc1, false, false);
    acc2 = __builtin_amdgcn_wmma_f32_16x16x32_f16(false, a, false, b2, (short)0, acc2, false, false);
    acc3 = __builtin_amdgcn_wmma_f32_16x16x32_f16(false, a, false, b3, (short)0, acc3, false, false);

    wait_async0();
    __syncthreads();
    buf ^= 1;
  }

  // C/D layout: VGPR v -> M = m0+v (lanes 0-15) / m0+v+8 (lanes 16-31); N = n0 + lane%16
  const int l  = lane & 15;
  const int hi = lane >> 4;
  const int m0 = m0b + wm * 16;
  v8f accs[4] = {acc0, acc1, acc2, acc3};
#pragma unroll
  for (int i = 0; i < 4; ++i) {
    const int n  = n0b + wn * 64 + i * 16 + l;
    const float bv = bias ? bias[n] : 0.f;
#pragma unroll
    for (int v = 0; v < 8; ++v) {
      const int r = m0 + v + hi * 8;
      float x = accs[i][v] + bv;
      if (act == 1)      x = act_elu(x);
      else if (act == 2) x = act_softplus01(x);
      if (outF16) outF16[(size_t)r * N + n] = (_Float16)x;
      else        outF32[(size_t)r * N + n] = x;
    }
  }
}

// ---------------------------------------------------------------------------
// One GRU timestep, fused GEMM + gates (direct loads; Whh is L2-resident).
// Each block owns 32 hidden columns j and computes the three gate tiles
// (j, j+CH, j+2*CH) of gh = h @ Whh^T itself -> no cross-block dependency.
// Block = 8 waves: 4 M-tiles (64 batch rows) x 2 column sub-tiles of 16.
// ---------------------------------------------------------------------------
__global__ void k_gru_step(const _Float16* __restrict__ hin,   // [64, CH]
                           _Float16*       __restrict__ hout,  // [64, CH]
                           const _Float16* __restrict__ Whh,   // [G3, CH]
                           const float*    __restrict__ bhh,   // [G3]
                           const _Float16* __restrict__ gx,    // [64, G3] this t
                           float*          __restrict__ hidden,// [64, CH] f32 out
                           _Float16*       __restrict__ hmu,   // [64, HENC]
                           _Float16*       __restrict__ hsig) {// [64, HENC]
  const int lane = threadIdx.x & 31;
  const int wave = threadIdx.x >> 5;
  const int wm   = wave & 3;
  const int wn   = wave >> 2;
  const int m0   = wm * 16;
  const int j0   = blockIdx.x * 32 + wn * 16;

  v8f ar = {}, az = {}, an = {};
  for (int k = 0; k < CH; k += 32) {
    v16h a  = load_frag16(hin, m0, CH, k, lane);
    v16h br = load_frag16(Whh, j0,          CH, k, lane);
    v16h bz = load_frag16(Whh, j0 + CH,     CH, k, lane);
    v16h bn = load_frag16(Whh, j0 + 2 * CH, CH, k, lane);
    ar = __builtin_amdgcn_wmma_f32_16x16x32_f16(false, a, false, br, (short)0, ar, false, false);
    az = __builtin_amdgcn_wmma_f32_16x16x32_f16(false, a, false, bz, (short)0, az, false, false);
    an = __builtin_amdgcn_wmma_f32_16x16x32_f16(false, a, false, bn, (short)0, an, false, false);
  }

  const int l  = lane & 15;
  const int hi = lane >> 4;
  const int j  = j0 + l;
  const float br_ = bhh[j];
  const float bz_ = bhh[j + CH];
  const float bn_ = bhh[j + 2 * CH];
#pragma unroll
  for (int v = 0; v < 8; ++v) {
    const int row = m0 + v + hi * 8;
    const size_t gbase = (size_t)row * G3 + j;
    const float gxr = (float)gx[gbase];
    const float gxz = (float)gx[gbase + CH];
    const float gxn = (float)gx[gbase + 2 * CH];
    const float r = sigmoid_(gxr + ar[v] + br_);
    const float z = sigmoid_(gxz + az[v] + bz_);
    const float n = tanhf(gxn + r * (an[v] + bn_));
    const float hp = (float)hin[(size_t)row * CH + j];
    const float hn = (1.f - z) * n + z * hp;
    hout[(size_t)row * CH + j]   = (_Float16)hn;
    hidden[(size_t)row * CH + j] = hn;
    if (j < HENC) hmu [(size_t)row * HENC + j]          = (_Float16)hn;
    else          hsig[(size_t)row * HENC + (j - HENC)] = (_Float16)hn;
  }
}

// ---------------------------------------------------------------------------
extern "C" void kernel_launch(void* const* d_in, const int* in_sizes, int n_in,
                              void* d_out, int out_size, void* d_ws, size_t ws_size,
                              hipStream_t stream) {
  (void)in_sizes; (void)n_in; (void)out_size; (void)ws_size;

  const float* x    = (const float*)d_in[0];
  const float* We   = (const float*)d_in[1];
  const float* be   = (const float*)d_in[2];
  const float* Wih  = (const float*)d_in[3];
  const float* bih  = (const float*)d_in[4];
  const float* Whh  = (const float*)d_in[5];
  const float* bhh  = (const float*)d_in[6];
  const float* Wmu  = (const float*)d_in[7];
  const float* bmu  = (const float*)d_in[8];
  const float* Wsig = (const float*)d_in[9];
  const float* bsig = (const float*)d_in[10];

  float* mu     = (float*)d_out;
  float* sig    = mu  + (size_t)TB * DOUT;
  float* hidden = sig + (size_t)TB * DOUT;

  // Workspace carve-out (f16 buffers, 256B aligned); total ~360 MB.
  char* ws = (char*)d_ws;
  size_t off = 0;
  auto take = [&](size_t elems) -> _Float16* {
    _Float16* p = (_Float16*)(ws + off);
    off += ((elems * sizeof(_Float16) + 255) & ~(size_t)255);
    return p;
  };
  _Float16* We16   = take((size_t)HENC * DIN);
  _Float16* Wih16  = take((size_t)G3 * HENC);
  _Float16* Whh16  = take((size_t)G3 * CH);
  _Float16* Wmu16  = take((size_t)DOUT * HENC);
  _Float16* Wsig16 = take((size_t)DOUT * HENC);
  _Float16* x16    = take((size_t)TB * DIN);
  _Float16* xe16   = take((size_t)TB * HENC);
  _Float16* gx16   = take((size_t)TB * G3);
  _Float16* hA     = take((size_t)BATCH * CH);
  _Float16* hB     = take((size_t)BATCH * CH);
  _Float16* hmu16  = take((size_t)TB * HENC);
  _Float16* hsig16 = take((size_t)TB * HENC);

  auto cvt = [&](const float* src, _Float16* dst, size_t n) {
    k_f32_to_f16<<<(unsigned)((n + 255) / 256), 256, 0, stream>>>(src, dst, (int)n);
  };
  cvt(x,    x16,    (size_t)TB * DIN);
  cvt(We,   We16,   (size_t)HENC * DIN);
  cvt(Wih,  Wih16,  (size_t)G3 * HENC);
  cvt(Whh,  Whh16,  (size_t)G3 * CH);
  cvt(Wmu,  Wmu16,  (size_t)DOUT * HENC);
  cvt(Wsig, Wsig16, (size_t)DOUT * HENC);

  // Encoder: xe = f16(ELU(x @ We^T + be))      [16384 x 1024], K=512
  k_gemm_wmma<<<dim3(HENC / BN, TB / BM), 256, 0, stream>>>(
      x16, We16, be, nullptr, xe16, TB, HENC, DIN, 1);

  // gx = f16(xe @ Wih^T + bih)                 [16384 x 6144], K=1024
  k_gemm_wmma<<<dim3(G3 / BN, TB / BM), 256, 0, stream>>>(
      xe16, Wih16, bih, nullptr, gx16, TB, G3, HENC, 0);

  // h0 = 0
  k_zero_f16<<<(BATCH * CH + 255) / 256, 256, 0, stream>>>(hA, BATCH * CH);

  // Sequential scan: 256 fused step kernels, ping-ponging h buffers.
  _Float16* hcur = hA;
  _Float16* hnxt = hB;
  for (int t = 0; t < T_STEPS; ++t) {
    k_gru_step<<<CH / 32, 256, 0, stream>>>(
        hcur, hnxt, Whh16, bhh,
        gx16   + (size_t)t * BATCH * G3,
        hidden + (size_t)t * BATCH * CH,
        hmu16  + (size_t)t * BATCH * HENC,
        hsig16 + (size_t)t * BATCH * HENC);
    _Float16* tmp = hcur; hcur = hnxt; hnxt = tmp;
  }

  // Heads: mu = ELU(h_mu @ Wmu^T + bmu), sig = softplus(h_sig @ Wsig^T + bsig) + 0.1
  k_gemm_wmma<<<dim3(DOUT / BN, TB / BM), 256, 0, stream>>>(
      hmu16, Wmu16, bmu, mu, nullptr, TB, DOUT, HENC, 1);
  k_gemm_wmma<<<dim3(DOUT / BN, TB / BM), 256, 0, stream>>>(
      hsig16, Wsig16, bsig, sig, nullptr, TB, DOUT, HENC, 2);
}
